// Encoder_33294586479284
// MI455X (gfx1250) — compile-verified, matrix operand order checked by reading, not run on hardware
//
#include <hip/hip_runtime.h>
#include <hip/hip_bf16.h>

typedef __attribute__((ext_vector_type(16))) _Float16 v16h;
typedef __attribute__((ext_vector_type(8)))  float    v8f;

#define B_    256
#define T_    128
#define N_    128
#define H_    256
#define K_HC  (2 * H_)    // 512
#define K_CAT (N_ + H_)   // 384
#define G4H   (4 * H_)    // 1024
#define MT    16          // batch tile (WMMA M)
#define NTHR  512         // 16 wave32s per WG

// K index pattern for 16-bit A/B fragments (ISA 7.12.2, 16-bit 16x32):
// lane half = lane>>4; VGPR j (0..7) holds K = (j/4)*16 + half*8 + (j%4)*2 + {0,1}
__host__ __device__ __forceinline__ int kpat(int j, int half) {
  return ((j >> 2) << 4) | (half << 3) | ((j & 3) << 1);
}

// A fragment from an LDS row-major [16][ld] f16 tile, k-window at k0.
// Per lane this is two contiguous 16B runs -> compiler emits 2x ds_load_b128.
__device__ __forceinline__ v16h load_a_lds(const _Float16* base, int ld, int row,
                                           int half, int k0) {
  v16h a;
#pragma unroll
  for (int j = 0; j < 8; ++j) {
    int k = k0 + kpat(j, half);
    a[2 * j]     = base[row * ld + k];
    a[2 * j + 1] = base[row * ld + k + 1];
  }
  return a;
}

// B fragment pre-packed in register layout: lane's 16 halves are contiguous (32B).
__device__ __forceinline__ v16h load_b_frag(const _Float16* __restrict__ p) {
  return *(const v16h*)p;
}

__device__ __forceinline__ float tanh_fast(float x) {
  float a = fabsf(x);
  float e = __expf(2.0f * a);
  float r = 1.0f - 2.0f / (e + 1.0f);
  return copysignf(r, x);
}
__device__ __forceinline__ float sigmoid_fast(float x) {
  return 1.0f / (1.0f + __expf(-x));
}

// ---- weight staging: f32 -> f16, transposed AND pre-shuffled into WMMA B-fragment
// register layout. Fragment f holds a 32(K) x 16(N) tile; element (L, e) of the
// fragment is B[k][n] with k = kt*32 + kpat(e>>1, L>>4) + (e&1), n = nt*16 + (L&15).
// Storage: out[f*512 + L*16 + e]  (so each lane's 16 halves are 32B-contiguous).

__global__ void pack_wah(const float* __restrict__ W_ah, _Float16* __restrict__ out) {
  int idx = blockIdx.x * blockDim.x + threadIdx.x;   // 128 frags * 512 = 65536
  int f = idx >> 9, r = idx & 511;
  int L = r >> 4, e = r & 15;
  int kt = f & 15, nt = f >> 4;                      // f = nt*16 + kt
  int k = kt * 32 + kpat(e >> 1, L >> 4) + (e & 1);  // hc dim (0..511)
  int n = nt * 16 + (L & 15);                        // T out dim (0..127)
  out[idx] = (_Float16)W_ah[n * K_HC + k];           // B = W_ah^T
}

__global__ void pack_wcat(const float* __restrict__ W_ih, const float* __restrict__ W_hh,
                          _Float16* __restrict__ out) {
  int idx = blockIdx.x * blockDim.x + threadIdx.x;   // 768 frags * 512 = 393216
  int f = idx >> 9, r = idx & 511;
  int L = r >> 4, e = r & 15;
  int kt = f % 12, nt = f / 12;                      // f = nt*12 + kt
  int k = kt * 32 + kpat(e >> 1, L >> 4) + (e & 1);  // concat dim (0..383)
  int n = nt * 16 + (L & 15);                        // 4H out dim (0..1023)
  float v = (k < N_) ? W_ih[n * N_ + k] : W_hh[n * H_ + (k - N_)];
  out[idx] = (_Float16)v;
}

// ---- persistent encoder: one WG owns a 16-row batch tile for the whole T loop ----
__global__ __launch_bounds__(NTHR) void encoder_kernel(
    const float* __restrict__ x,      // [B,T,N]
    const float* __restrict__ b_ah,   // [T]
    const float* __restrict__ W_ai,   // [T]
    const float* __restrict__ b_ai,   // [1]
    const float* __restrict__ W_a,    // [T]
    const float* __restrict__ b_a,    // [1]
    const float* __restrict__ b_ih,   // [4H]
    const float* __restrict__ b_hh,   // [4H]
    const _Float16* __restrict__ WahP,   // 128 frags * 512 halves
    const _Float16* __restrict__ WcatP,  // 768 frags * 512 halves
    float* __restrict__ d_iw,   // [B,T,N]
    float* __restrict__ d_ie)   // [B,T,H]
{
  __shared__ float    h_s[MT][H_];          // 16 KB
  __shared__ float    c_s[MT][H_];          // 16 KB
  __shared__ _Float16 abuf[MT * K_HC];      // 16 KB: hc16 [16][512] reused as wxh16 [16][384]
  __shared__ float    hpart_s[MT][T_ + 1];  // padded stride: kill 16-way bank conflict
  __shared__ float    ipart_s[MT][N_];      // 8 KB
  __shared__ float    score_s[MT][N_];      // 8 KB
  __shared__ float    gates_s[MT][G4H];     // 64 KB
  __shared__ float    wa_s[T_];

  const int tid  = threadIdx.x;
  const int lane = tid & 31;   // wave32
  const int wv   = tid >> 5;   // 16 waves
  const int half = lane >> 4;
  const int l16  = lane & 15;
  const int bg   = blockIdx.x * MT;

  if (tid < T_) wa_s[tid] = W_a[tid];
  for (int i = tid; i < MT * H_; i += NTHR) { h_s[0][i] = 0.f; c_s[0][i] = 0.f; }

  // ipart[b][n] = sum_t x[b,t,n] * W_ai[t] + b_ai  (time-invariant); 4 per thread
  {
    const float bai = b_ai[0];
    int b  = tid & 15;
    int n0 = (tid >> 4) * 4;
    const float* xb = x + (size_t)(bg + b) * T_ * N_;
#pragma unroll 1
    for (int q = 0; q < 4; ++q) {
      int n = n0 + q;
      float acc = 0.f;
      for (int tt = 0; tt < T_; ++tt) acc += xb[tt * N_ + n] * W_ai[tt];
      ipart_s[b][n] = acc + bai;
    }
  }
  __syncthreads();

  const float ba0 = b_a[0];

  for (int t = 0; t < T_; ++t) {
    // phase 1: stage hc = [h, c] as f16 in LDS
    for (int i = tid; i < MT * K_HC; i += NTHR) {
      int b = i >> 9, k = i & 511;
      abuf[i] = (_Float16)((k < H_) ? h_s[b][k] : c_s[b][k - H_]);
    }
    __syncthreads();

    // phase 2: hpart = hc16 @ Wah^T + b_ah ; waves 0..7 -> one 16x16 n-tile each
    if (wv < 8) {
      v8f acc = {};
      int n0 = wv * 16;
#pragma unroll 2
      for (int kt = 0; kt < K_HC / 32; ++kt) {
        v16h a = load_a_lds(abuf, K_HC, l16, half, kt * 32);
        v16h b = load_b_frag(WahP + ((size_t)(wv * 16 + kt) * 32 + lane) * 16);
        acc = __builtin_amdgcn_wmma_f32_16x16x32_f16(false, a, false, b,
                                                     (short)0, acc, false, false);
      }
      int n = n0 + l16;
      float bias = b_ah[n];
      int r0 = half * 8;
#pragma unroll
      for (int i = 0; i < 8; ++i) hpart_s[r0 + i][n] = acc[i] + bias;
    }
    __syncthreads();

    // phase 3: score[b][n] = b_a + sum_t wa[t]*tanh(hpart[b][t] + ipart[b][n]);
    //          4 outputs per thread, hpart load hoisted per t.
    //          Also refill abuf as wxh16 [16][384] h-part (hc16 dead now).
    {
      int b  = tid & 15;
      int n0 = (tid >> 4) * 4;
      float ip[4], acc[4] = {0.f, 0.f, 0.f, 0.f};
#pragma unroll
      for (int q = 0; q < 4; ++q) ip[q] = ipart_s[b][n0 + q];
#pragma unroll 1
      for (int tt = 0; tt < T_; ++tt) {
        float ph = hpart_s[b][tt];
        float wa = wa_s[tt];
#pragma unroll
        for (int q = 0; q < 4; ++q) acc[q] += wa * tanh_fast(ph + ip[q]);
      }
#pragma unroll
      for (int q = 0; q < 4; ++q) score_s[b][n0 + q] = acc[q] + ba0;

      for (int i = tid; i < MT * H_; i += NTHR) {
        int b2 = i >> 8, k = i & 255;
        abuf[b2 * K_CAT + N_ + k] = (_Float16)h_s[b2][k];
      }
    }
    __syncthreads();

    // phase 4: softmax over N + weighted input; one wave per batch row,
    //          wave32 shuffle reductions.
    {
      int b = wv;
      float sc[4];
      float m = -1e30f;
#pragma unroll
      for (int i = 0; i < 4; ++i) { sc[i] = score_s[b][lane + 32 * i]; m = fmaxf(m, sc[i]); }
#pragma unroll
      for (int off = 16; off > 0; off >>= 1) m = fmaxf(m, __shfl_xor(m, off, 32));
      float e[4], s = 0.f;
#pragma unroll
      for (int i = 0; i < 4; ++i) { e[i] = __expf(sc[i] - m); s += e[i]; }
#pragma unroll
      for (int off = 16; off > 0; off >>= 1) s += __shfl_xor(s, off, 32);
      float inv = 1.0f / s;
      const float* xr = x    + ((size_t)(bg + b) * T_ + t) * N_;
      float*       ow = d_iw + ((size_t)(bg + b) * T_ + t) * N_;
#pragma unroll
      for (int i = 0; i < 4; ++i) {
        int n = lane + 32 * i;
        float wx = e[i] * inv * xr[n];
        ow[n] = wx;
        abuf[b * K_CAT + n] = (_Float16)wx;
      }
    }
    __syncthreads();

    // phase 5: gates = wxh16 @ [W_ih;W_hh]^T + b_ih + b_hh ; 16 waves x 4 n-tiles
#pragma unroll 1
    for (int tt = 0; tt < 4; ++tt) {
      int g  = wv * 4 + tt;      // 0..63
      int n0 = g * 16;
      v8f acc = {};
#pragma unroll 2
      for (int kt = 0; kt < K_CAT / 32; ++kt) {
        v16h a = load_a_lds(abuf, K_CAT, l16, half, kt * 32);
        v16h b = load_b_frag(WcatP + ((size_t)(g * 12 + kt) * 32 + lane) * 16);
        acc = __builtin_amdgcn_wmma_f32_16x16x32_f16(false, a, false, b,
                                                     (short)0, acc, false, false);
      }
      int n = n0 + l16;
      float bias = b_ih[n] + b_hh[n];
      int r0 = half * 8;
#pragma unroll
      for (int i = 0; i < 8; ++i) gates_s[r0 + i][n] = acc[i] + bias;
    }
    __syncthreads();

    // phase 6: LSTM pointwise, update h/c in LDS, emit encoded row
#pragma unroll 1
    for (int r = 0; r < (MT * H_) / NTHR; ++r) {
      int o = tid + r * NTHR;
      int b = o >> 8, k = o & 255;
      float ig = sigmoid_fast(gates_s[b][k]);
      float fg = sigmoid_fast(gates_s[b][H_ + k]);
      float gg = tanh_fast(gates_s[b][2 * H_ + k]);
      float og = sigmoid_fast(gates_s[b][3 * H_ + k]);
      float c2 = fg * c_s[b][k] + ig * gg;
      float h2 = og * tanh_fast(c2);
      c_s[b][k] = c2;
      h_s[b][k] = h2;
      d_ie[((size_t)(bg + b) * T_ + t) * H_ + k] = h2;
    }
    __syncthreads();
  }
}

extern "C" void kernel_launch(void* const* d_in, const int* in_sizes, int n_in,
                              void* d_out, int out_size, void* d_ws, size_t ws_size,
                              hipStream_t stream) {
  const float* x    = (const float*)d_in[0];
  const float* W_ah = (const float*)d_in[1];
  const float* b_ah = (const float*)d_in[2];
  const float* W_ai = (const float*)d_in[3];
  const float* b_ai = (const float*)d_in[4];
  const float* W_a  = (const float*)d_in[5];
  const float* b_a  = (const float*)d_in[6];
  const float* W_ih = (const float*)d_in[7];
  const float* W_hh = (const float*)d_in[8];
  const float* b_ih = (const float*)d_in[9];
  const float* b_hh = (const float*)d_in[10];

  // workspace: pre-shuffled f16 B-fragments (~0.9 MB, L2-resident)
  _Float16* WahP  = (_Float16*)d_ws;                                   // 128*512 halves
  _Float16* WcatP = (_Float16*)((char*)d_ws + (size_t)128 * 512 * 2);  // 768*512 halves

  pack_wah <<<(128 * 512) / 256, 256, 0, stream>>>(W_ah, WahP);
  pack_wcat<<<(768 * 512) / 256, 256, 0, stream>>>(W_ih, W_hh, WcatP);

  float* d_iw = (float*)d_out;
  float* d_ie = d_iw + (size_t)B_ * T_ * N_;

  encoder_kernel<<<B_ / MT, NTHR, 0, stream>>>(x, b_ah, W_ai, b_ai, W_a, b_a,
                                               b_ih, b_hh, WahP, WcatP, d_iw, d_ie);
}
